// TTS_56281251447323
// MI455X (gfx1250) — matrix operand order repaired and checked by reading, not compiled, not run on hardware
//
#include <hip/hip_runtime.h>
#include <cstdint>
#include <cstddef>

typedef __attribute__((ext_vector_type(16))) _Float16 v16h;
typedef __attribute__((ext_vector_type(8)))  _Float16 v8h;
typedef __attribute__((ext_vector_type(4)))  _Float16 v4h;
typedef __attribute__((ext_vector_type(8)))  float    v8f;

#define B_      128
#define TIN_    512
#define D_      256
#define MEL_    80
#define RR_     5
#define STEPS_  400
#define STY_    10
#define KP_     96      // indic K padded 80 -> 96

union AFrag { v16h v; v8h g[2]; _Float16 h[16]; };
union CFrag { v8f  v; float f[8]; };

__device__ __forceinline__ float sigf(float x) { return 1.f / (1.f + expf(-x)); }

// ---------------------------------------------------------------------------
// WMMA GEMM:  C[M,N] = act(A[M,K](f16) * B[K,N] + bias)
// B pre-packed in WMMA fragment order:
//   Bp[((ct*ktiles + kc)*32 + lane)*16 + j] = B[kc*32+(lane>>4)*8+(j&7)+(j>>3)*16][ct*16+(lane&15)]
// Block 256 thr = 8 waves, tile 32(M) x 64(N), K chunks of 32.
// GUARD=false: M%32==0, K==lda==ktiles*32, A is f16; stage A via
//   GLOBAL_LOAD_ASYNC_TO_LDS_B64 with double-buffered LDS (1 barrier/chunk).
// GUARD=true: A is f32 with full bounds checks (tiny keys_s GEMM only).
// ---------------------------------------------------------------------------
template <bool OUT16, int ACT, bool GUARD>
__global__ void tts_gemm(const void* __restrict__ A, int lda,
                         const _Float16* __restrict__ Bp, int ktiles,
                         const float* __restrict__ bias,
                         void* __restrict__ C, int ldc,
                         int M, int N, int K)
{
    __shared__ __align__(16) _Float16 As[2][32][40];   // row stride 80B

    const int tid  = threadIdx.x;
    const int wave = tid >> 5;
    const int lane = tid & 31;
    const int row0 = blockIdx.y * 32;
    const int col0 = blockIdx.x * 64;
    const int tm   = (wave >> 2) * 16;
    const int tn   = (wave & 3) * 16;
    const int ct   = (col0 + tn) >> 4;

    const _Float16* bptr = Bp + (((size_t)ct * ktiles) * 32 + lane) * 16;
    const int m  = lane & 15;
    const int kb = (lane >> 4) * 8;

    CFrag acc;
    acc.v = (v8f){};

    if (!GUARD) {
        // -------- async double-buffered pipeline --------
        const int r = tid >> 3;              // 0..31
        const int c = (tid & 7) * 4;         // 0..28
        const _Float16* a0 = (const _Float16*)A + (size_t)(row0 + r) * lda + c;
        const uint32_t lds0 = (uint32_t)(uintptr_t)&As[0][r][c];
        const uint32_t lds1 = (uint32_t)(uintptr_t)&As[1][r][c];

        asm volatile("global_load_async_to_lds_b64 %0, %1, off"
                     :: "v"(lds0), "v"(a0) : "memory");

        for (int kc = 0; kc < ktiles; ++kc) {
            asm volatile("s_wait_asynccnt 0x0" ::: "memory");
            __syncthreads();                 // chunk kc visible; prev buffer free
            if (kc + 1 < ktiles) {
                const uint32_t dst = ((kc + 1) & 1) ? lds1 : lds0;
                const _Float16* an = a0 + (size_t)(kc + 1) * 32;
                asm volatile("global_load_async_to_lds_b64 %0, %1, off"
                             :: "v"(dst), "v"(an) : "memory");
            }
            const _Float16 (*Ab)[40] = As[kc & 1];
            AFrag af;
            af.g[0] = *(const v8h*)&Ab[tm + m][kb];
            af.g[1] = *(const v8h*)&Ab[tm + m][16 + kb];
            const v16h bv = *(const v16h*)bptr;
            bptr += 32 * 16;
            acc.v = __builtin_amdgcn_wmma_f32_16x16x32_f16(
                false, af.v, false, bv, (short)0, acc.v, false, false);
        }
    } else {
        // -------- guarded synchronous path (f32 A) --------
        for (int kc = 0; kc < ktiles; ++kc) {
            const int r  = tid >> 3;
            const int c  = (tid & 7) * 4;
            const int rr = row0 + r;
#pragma unroll
            for (int i = 0; i < 4; ++i) {
                const int kk = kc * 32 + c + i;
                float v = 0.f;
                if (rr < M && kk < K) v = ((const float*)A)[(size_t)rr * lda + kk];
                As[0][r][c + i] = (_Float16)v;
            }
            __syncthreads();
            AFrag af;
            af.g[0] = *(const v8h*)&As[0][tm + m][kb];
            af.g[1] = *(const v8h*)&As[0][tm + m][16 + kb];
            const v16h bv = *(const v16h*)bptr;
            bptr += 32 * 16;
            acc.v = __builtin_amdgcn_wmma_f32_16x16x32_f16(
                false, af.v, false, bv, (short)0, acc.v, false, false);
            __syncthreads();
        }
    }

    // ---- store ----
    const int n   = lane & 15;
    const int col = col0 + tn + n;
    if (col < N) {
        const float bvs = bias ? bias[col] : 0.f;
#pragma unroll
        for (int r = 0; r < 8; ++r) {
            const int mm  = r + (lane >> 4) * 8;
            const int row = row0 + tm + mm;
            if (row < M) {
                float v = acc.f[r] + bvs;
                if (ACT == 1) v = fmaxf(v, 0.f);
                if (OUT16) ((_Float16*)C)[(size_t)row * ldc + col] = (_Float16)v;
                else       ((float*)C)[(size_t)row * ldc + col] = v;
            }
        }
    }
}

// ---------------------------------------------------------------------------
// Pack a K x N f32 weight into WMMA B-fragment order f16 (zero-padded).
// ---------------------------------------------------------------------------
__global__ void tts_pack_b(const float* __restrict__ src, _Float16* __restrict__ dst,
                           int K, int N, int ntpad, int ktiles)
{
    const int idx = blockIdx.x * blockDim.x + threadIdx.x;
    const int total = ntpad * ktiles * 512;
    if (idx >= total) return;
    const int j    = idx & 15;
    const int t1   = idx >> 4;
    const int lane = t1 & 31;
    const int t2   = t1 >> 5;
    const int kc   = t2 % ktiles;
    const int ctl  = t2 / ktiles;
    const int n = ctl * 16 + (lane & 15);
    const int k = kc * 32 + (lane >> 4) * 8 + (j & 7) + (j >> 3) * 16;
    float v = 0.f;
    if (k < K && n < N) v = src[(size_t)k * N + n];
    dst[idx] = (_Float16)v;
}

// flat f32 -> f16
__global__ void tts_conv(const float* __restrict__ s, _Float16* __restrict__ d, int n)
{
    const int i = blockIdx.x * blockDim.x + threadIdx.x;
    if (i < n) d[i] = (_Float16)s[i];
}

// indic rows ordered [t*128 + b], f16, width KP_ (80 real + 16 zero pad)
__global__ void tts_build_indic(const float* __restrict__ mel, _Float16* __restrict__ indic)
{
    const int idx = blockIdx.x * blockDim.x + threadIdx.x;
    if (idx >= STEPS_ * B_ * KP_) return;
    const int m   = idx % KP_;
    const int row = idx / KP_;
    const int b   = row % B_;
    const int t   = row / B_;
    float v = 0.f;
    if (t > 0 && m < MEL_)
        v = mel[((size_t)b * (STEPS_ * RR_) + (t * RR_ - 1)) * MEL_ + m];
    indic[idx] = (_Float16)v;
}

// ---------------------------------------------------------------------------
// GRU combine: h' = z*h + (1-z)*tanh(nx + r*nh)
// xp (f16) = [zx|rx|nx] with bias folded; hW (f32) = [zh|rh|nh]
// Writes f32 state h + f16 mirror h16; optional o16 = f16(h' + res16).
// ---------------------------------------------------------------------------
__global__ void tts_gru_combine(const _Float16* __restrict__ xp,
                                const float* __restrict__ hW,
                                float* __restrict__ h,
                                _Float16* __restrict__ h16,
                                _Float16* __restrict__ o16,
                                const _Float16* __restrict__ res16)
{
    const int idx = blockIdx.x * blockDim.x + threadIdx.x;  // 0 .. B_*D_-1
    const int b = idx >> 8, d = idx & 255;
    const int base = b * 768;
    const float zx = (float)xp[base + d];
    const float rx = (float)xp[base + 256 + d];
    const float nx = (float)xp[base + 512 + d];
    const float zh = hW[base + d], rh = hW[base + 256 + d], nh = hW[base + 512 + d];
    const float hv = h[idx];
    const float z  = sigf(zx + zh);
    const float r  = sigf(rx + rh);
    const float n  = tanhf(nx + r * nh);
    const float hn = z * hv + (1.f - z) * n;
    h[idx]   = hn;
    h16[idx] = (_Float16)hn;
    if (o16) o16[idx] = (_Float16)(hn + (float)res16[idx]);
}

// ---------------------------------------------------------------------------
// Content attention: one block (8 waves) per batch row.
// ---------------------------------------------------------------------------
__global__ void tts_attn(const _Float16* __restrict__ keys16,
                         const float* __restrict__ enc,
                         const float* __restrict__ qcat,
                         const float* __restrict__ v_att,
                         const int* __restrict__ inp_mask,
                         float* __restrict__ context,
                         float* __restrict__ alpha_out)
{
    const int b = blockIdx.x, tid = threadIdx.x;
    const int wv = tid >> 5, ln = tid & 31;
    __shared__ float ealpha[TIN_], red[256];

    float q8[8], w8[8];
#pragma unroll
    for (int i = 0; i < 8; ++i) {
        q8[i] = qcat[b * 2 * D_ + ln * 8 + i];
        w8[i] = v_att[ln * 8 + i];
    }
    const int Tv = inp_mask[b];

    for (int t = wv; t < TIN_; t += 8) {
        const v8h kv = *(const v8h*)(keys16 + ((size_t)b * TIN_ + t) * D_ + ln * 8);
        float s = 0.f;
#pragma unroll
        for (int i = 0; i < 8; ++i)
            s += tanhf((float)kv[i] + q8[i]) * w8[i];
#pragma unroll
        for (int off = 16; off > 0; off >>= 1) s += __shfl_down(s, off, 32);
        if (ln == 0) ealpha[t] = (t < Tv) ? s : -1e9f;
    }
    __syncthreads();

    const float e0 = ealpha[tid], e1 = ealpha[tid + 256];
    red[tid] = fmaxf(e0, e1);
    __syncthreads();
    for (int off = 128; off > 0; off >>= 1) {
        if (tid < off) red[tid] = fmaxf(red[tid], red[tid + off]);
        __syncthreads();
    }
    const float mx = red[0];
    __syncthreads();
    const float x0 = expf(e0 - mx), x1 = expf(e1 - mx);
    red[tid] = x0 + x1;
    __syncthreads();
    for (int off = 128; off > 0; off >>= 1) {
        if (tid < off) red[tid] += red[tid + off];
        __syncthreads();
    }
    const float inv = 1.f / red[0];
    __syncthreads();
    const float a0 = x0 * inv, a1 = x1 * inv;
    ealpha[tid] = a0; ealpha[tid + 256] = a1;
    alpha_out[(size_t)b * (STEPS_ * TIN_) + tid]       = a0;
    alpha_out[(size_t)b * (STEPS_ * TIN_) + tid + 256] = a1;
    __syncthreads();

    float c = 0.f;
    for (int t = 0; t < TIN_; ++t)
        c = fmaf(ealpha[t], enc[((size_t)b * TIN_ + t) * D_ + tid], c);
    context[b * D_ + tid] = c;
}

// ---------------------------------------------------------------------------
// Style attention + w + wc; builds cat16 = f16([att_h | wc]).
// ---------------------------------------------------------------------------
__global__ void tts_style(const float* __restrict__ keys_s,
                          const float* __restrict__ qcat,
                          const float* __restrict__ v_s,
                          const float* __restrict__ stok,
                          const float* __restrict__ Ww,
                          const float* __restrict__ bw,
                          const float* __restrict__ att_h,
                          const float* __restrict__ context,
                          _Float16* __restrict__ cat16,
                          float* __restrict__ alphas_out,
                          float* __restrict__ w_out)
{
    const int b = blockIdx.x, tid = threadIdx.x;
    __shared__ float red[256];
    __shared__ float es[STY_];
    __shared__ float wsh;
    const float q = qcat[b * 2 * D_ + 256 + tid];

    for (int s = 0; s < STY_; ++s) {
        red[tid] = tanhf(keys_s[s * D_ + tid] + q) * v_s[tid];
        __syncthreads();
        for (int off = 128; off > 0; off >>= 1) {
            if (tid < off) red[tid] += red[tid + off];
            __syncthreads();
        }
        if (tid == 0) es[s] = red[0];
        __syncthreads();
    }
    red[tid] = att_h[b * D_ + tid] * Ww[tid];
    __syncthreads();
    for (int off = 128; off > 0; off >>= 1) {
        if (tid < off) red[tid] += red[tid + off];
        __syncthreads();
    }
    if (tid == 0) wsh = 1.f / (1.f + expf(-(red[0] + bw[0])));
    __syncthreads();

    float mx = es[0];
#pragma unroll
    for (int s = 1; s < STY_; ++s) mx = fmaxf(mx, es[s]);
    float sum = 0.f;
#pragma unroll
    for (int s = 0; s < STY_; ++s) sum += expf(es[s] - mx);
    const float inv = 1.f / sum;
    float cs = 0.f;
#pragma unroll
    for (int s = 0; s < STY_; ++s) cs += expf(es[s] - mx) * inv * stok[s * D_ + tid];

    const float w  = wsh;
    const float wc = context[b * D_ + tid] + w * tanhf(cs);
    cat16[b * 512 + tid]       = (_Float16)att_h[b * D_ + tid];
    cat16[b * 512 + 256 + tid] = (_Float16)wc;
    if (tid < STY_) alphas_out[(size_t)b * (STEPS_ * STY_) + tid] = expf(es[tid] - mx) * inv;
    if (tid == 0)   w_out[(size_t)b * STEPS_] = w;
}

// ---------------------------------------------------------------------------
// Host launch
// ---------------------------------------------------------------------------
static inline dim3 gg(int M, int N) { return dim3((N + 63) / 64, (M + 31) / 32); }

extern "C" void kernel_launch(void* const* d_in, const int* in_sizes, int n_in,
                              void* d_out, int out_size, void* d_ws, size_t ws_size,
                              hipStream_t stream)
{
    (void)in_sizes; (void)n_in; (void)out_size; (void)ws_size;

    const float* enc   = (const float*)d_in[0];
    const int*   mask  = (const int*)d_in[1];
    const float* stok  = (const float*)d_in[2];
    const float* mel   = (const float*)d_in[3];
    const float* Wp1   = (const float*)d_in[4];
    const float* bp1   = (const float*)d_in[5];
    const float* Wp2   = (const float*)d_in[6];
    const float* bp2   = (const float*)d_in[7];
    const float* WxA   = (const float*)d_in[8];
    const float* WhA   = (const float*)d_in[9];
    const float* bA    = (const float*)d_in[10];
    const float* Wk    = (const float*)d_in[11];
    const float* Wq    = (const float*)d_in[12];
    const float* vatt  = (const float*)d_in[13];
    const float* Wks   = (const float*)d_in[14];
    const float* Wqs   = (const float*)d_in[15];
    const float* vs    = (const float*)d_in[16];
    const float* Ww    = (const float*)d_in[17];
    const float* bw    = (const float*)d_in[18];
    const float* Wa    = (const float*)d_in[19];
    const float* ba    = (const float*)d_in[20];
    const float* Wx1   = (const float*)d_in[21];
    const float* Wh1   = (const float*)d_in[22];
    const float* b1    = (const float*)d_in[23];
    const float* Wx2   = (const float*)d_in[24];
    const float* Wh2   = (const float*)d_in[25];
    const float* b2    = (const float*)d_in[26];
    const float* Wo    = (const float*)d_in[27];
    const float* bo    = (const float*)d_in[28];
    float* out = (float*)d_out;

    const size_t alpha_base  = (size_t)B_ * STEPS_ * RR_ * MEL_;
    const size_t alphas_base = alpha_base + (size_t)B_ * STEPS_ * TIN_;
    const size_t w_base      = alphas_base + (size_t)B_ * STEPS_ * STY_;

    // --- workspace carve ---
    char* p = (char*)d_ws;
    auto ah = [&](size_t n) -> _Float16* { _Float16* r = (_Float16*)p; p += ((n * 2 + 255) / 256) * 256; return r; };
    auto af = [&](size_t n) -> float*    { float*    r = (float*)p;    p += ((n * 4 + 255) / 256) * 256; return r; };
    auto psz = [](int ntpad, int ktiles) -> size_t { return (size_t)ntpad * ktiles * 512; };

    _Float16* pWp1 = ah(psz(16, 3));
    _Float16* pWp2 = ah(psz(8, 8));
    _Float16* pWxA = ah(psz(48, 4));
    _Float16* pWhA = ah(psz(48, 8));
    _Float16* pWk  = ah(psz(16, 8));
    _Float16* pWqc = ah(psz(32, 8));
    _Float16* pWks = ah(psz(16, 8));
    _Float16* pWa  = ah(psz(16, 16));
    _Float16* pWx1 = ah(psz(48, 8));
    _Float16* pWh1 = ah(psz(48, 8));
    _Float16* pWx2 = ah(psz(48, 8));
    _Float16* pWh2 = ah(psz(48, 8));
    _Float16* pWo  = ah(psz(28, 8));
    _Float16* indic = ah((size_t)STEPS_ * B_ * KP_);
    _Float16* P1    = ah((size_t)STEPS_ * B_ * 256);
    _Float16* P2    = ah((size_t)STEPS_ * B_ * 128);
    _Float16* xatt  = ah((size_t)STEPS_ * B_ * 768);
    _Float16* keys  = ah((size_t)B_ * TIN_ * D_);
    _Float16* enc16 = ah((size_t)B_ * TIN_ * D_);
    _Float16* attH16 = ah(B_ * D_);
    _Float16* h1_16  = ah(B_ * D_);
    _Float16* h2_16  = ah(B_ * D_);
    _Float16* aco16  = ah(B_ * D_);
    _Float16* o1_16  = ah(B_ * D_);
    _Float16* o2_16  = ah(B_ * D_);
    _Float16* cat16  = ah(B_ * 512);
    _Float16* xg16   = ah(B_ * 768);
    float* keysS = af(STY_ * D_);
    float* attH  = af(B_ * D_);
    float* h1    = af(B_ * D_);
    float* h2    = af(B_ * D_);
    float* qcat  = af(B_ * 512);
    float* hWb   = af(B_ * 768);
    float* ctx   = af(B_ * D_);

    hipMemsetAsync(attH, 0, B_ * D_ * sizeof(float), stream);
    hipMemsetAsync(h1,   0, B_ * D_ * sizeof(float), stream);
    hipMemsetAsync(h2,   0, B_ * D_ * sizeof(float), stream);
    hipMemsetAsync(attH16, 0, B_ * D_ * sizeof(_Float16), stream);
    hipMemsetAsync(h1_16,  0, B_ * D_ * sizeof(_Float16), stream);
    hipMemsetAsync(h2_16,  0, B_ * D_ * sizeof(_Float16), stream);

    // --- pack weights into WMMA fragment order ---
    auto pk = [&](const float* s, _Float16* d, int K, int N, int ntpad, int kt) {
        const int tot = ntpad * kt * 512;
        tts_pack_b<<<(tot + 255) / 256, 256, 0, stream>>>(s, d, K, N, ntpad, kt);
    };
    pk(Wp1, pWp1, 80, 256, 16, 3);
    pk(Wp2, pWp2, 256, 128, 8, 8);
    pk(WxA, pWxA, 128, 768, 48, 4);
    pk(WhA, pWhA, 256, 768, 48, 8);
    pk(Wk,  pWk,  256, 256, 16, 8);
    pk(Wq,  pWqc, 256, 256, 16, 8);
    pk(Wqs, pWqc + psz(16, 8), 256, 256, 16, 8);
    pk(Wks, pWks, 256, 256, 16, 8);
    pk(Wa,  pWa,  512, 256, 16, 16);
    pk(Wx1, pWx1, 256, 768, 48, 8);
    pk(Wh1, pWh1, 256, 768, 48, 8);
    pk(Wx2, pWx2, 256, 768, 48, 8);
    pk(Wh2, pWh2, 256, 768, 48, 8);
    pk(Wo,  pWo,  256, 400, 28, 8);

    // --- precompute ---
    tts_build_indic<<<(STEPS_ * B_ * KP_ + 255) / 256, 256, 0, stream>>>(mel, indic);
    {
        const int n = B_ * TIN_ * D_;
        tts_conv<<<(n + 255) / 256, 256, 0, stream>>>(enc, enc16, n);
    }

    const int MT = STEPS_ * B_;  // 51200
    tts_gemm<true, 1, false><<<gg(MT, 256), 256, 0, stream>>>(indic, KP_, pWp1, 3, bp1, P1, 256, MT, 256, KP_);
    tts_gemm<true, 1, false><<<gg(MT, 128), 256, 0, stream>>>(P1, 256, pWp2, 8, bp2, P2, 128, MT, 128, 256);
    tts_gemm<true, 0, false><<<gg(MT, 768), 256, 0, stream>>>(P2, 128, pWxA, 4, bA, xatt, 768, MT, 768, 128);
    tts_gemm<true, 0, false><<<gg(B_ * TIN_, 256), 256, 0, stream>>>(enc16, D_, pWk, 8, nullptr, keys, 256, B_ * TIN_, 256, 256);
    tts_gemm<false, 0, true><<<gg(STY_, 256), 256, 0, stream>>>(stok, D_, pWks, 8, nullptr, keysS, 256, STY_, 256, 256);

    // --- recurrent loop ---
    for (int t = 0; t < STEPS_; ++t) {
        tts_gemm<false, 0, false><<<gg(B_, 768), 256, 0, stream>>>(attH16, D_, pWhA, 8, nullptr, hWb, 768, B_, 768, 256);
        tts_gru_combine<<<B_, 256, 0, stream>>>(xatt + (size_t)t * B_ * 768, hWb, attH, attH16, nullptr, nullptr);

        tts_gemm<false, 0, false><<<gg(B_, 512), 256, 0, stream>>>(attH16, D_, pWqc, 8, nullptr, qcat, 512, B_, 512, 256);

        tts_attn<<<B_, 256, 0, stream>>>(keys, enc, qcat, vatt, mask, ctx,
                                         out + alpha_base + (size_t)t * TIN_);
        tts_style<<<B_, 256, 0, stream>>>(keysS, qcat, vs, stok, Ww, bw, attH, ctx, cat16,
                                          out + alphas_base + (size_t)t * STY_,
                                          out + w_base + t);

        tts_gemm<true, 0, false><<<gg(B_, 256), 256, 0, stream>>>(cat16, 512, pWa, 16, ba, aco16, 256, B_, 256, 512);

        tts_gemm<true, 0, false><<<gg(B_, 768), 256, 0, stream>>>(aco16, D_, pWx1, 8, b1, xg16, 768, B_, 768, 256);
        tts_gemm<false, 0, false><<<gg(B_, 768), 256, 0, stream>>>(h1_16, D_, pWh1, 8, nullptr, hWb, 768, B_, 768, 256);
        tts_gru_combine<<<B_, 256, 0, stream>>>(xg16, hWb, h1, h1_16, o1_16, aco16);

        tts_gemm<true, 0, false><<<gg(B_, 768), 256, 0, stream>>>(o1_16, D_, pWx2, 8, b2, xg16, 768, B_, 768, 256);
        tts_gemm<false, 0, false><<<gg(B_, 768), 256, 0, stream>>>(h2_16, D_, pWh2, 8, nullptr, hWb, 768, B_, 768, 256);
        tts_gru_combine<<<B_, 256, 0, stream>>>(xg16, hWb, h2, h2_16, o2_16, o1_16);

        tts_gemm<false, 0, false><<<gg(B_, 400), 256, 0, stream>>>(
            o2_16, D_, pWo, 8, bo,
            out + (size_t)t * RR_ * MEL_, STEPS_ * RR_ * MEL_,
            B_, RR_ * MEL_, 256);
    }
}
